// MarginalRankingLoss_70669391888899
// MI455X (gfx1250) — matrix operand order repaired and checked by reading, not compiled, not run on hardware
//
#include <hip/hip_runtime.h>

#define DIMS   256
#define TK     50
#define NPART  32
#define MARGIN_C 0.9f

typedef __bf16 v16bf __attribute__((ext_vector_type(16)));
typedef float  v8f   __attribute__((ext_vector_type(8)));
typedef unsigned int u32x4 __attribute__((ext_vector_type(4)));

// round-to-nearest-even f32 -> bf16, packed pair into one dword
__device__ __forceinline__ unsigned int pack2bf(float a, float b) {
    union { float f; unsigned int u; } ua, ub;
    ua.f = a; ub.f = b;
    unsigned int ra = (ua.u + 0x7FFFu + ((ua.u >> 16) & 1u)) >> 16;
    unsigned int rb = (ub.u + 0x7FFFu + ((ub.u >> 16) & 1u)) >> 16;
    return (ra & 0xFFFFu) | ((rb & 0xFFFFu) << 16);
}

// --------------------------------------------------------------------------
// K1: per-row inverse L2 norm (f32) + bf16 copy.  One wave per row.
// --------------------------------------------------------------------------
__global__ __launch_bounds__(256) void k_norm_convert(
    const float* __restrict__ X, unsigned short* __restrict__ Xb,
    float* __restrict__ rn, int nrows, int npad)
{
    int row  = blockIdx.x * 8 + (threadIdx.x >> 5);
    int lane = threadIdx.x & 31;
    if (row >= npad) return;
    u32x4* dst = (u32x4*)(Xb + (size_t)row * DIMS + lane * 8);
    if (row >= nrows) {
        u32x4 z = {0u, 0u, 0u, 0u};
        *dst = z;
        if (lane == 0) rn[row] = 0.0f;
        return;
    }
    const float* src = X + (size_t)row * DIMS + lane * 8;
    float f[8];
    #pragma unroll
    for (int i = 0; i < 8; ++i) f[i] = src[i];
    float ss = 0.0f;
    #pragma unroll
    for (int i = 0; i < 8; ++i) ss += f[i] * f[i];
    #pragma unroll
    for (int m = 16; m >= 1; m >>= 1) ss += __shfl_xor(ss, m, 32);
    u32x4 w;
    w[0] = pack2bf(f[0], f[1]);
    w[1] = pack2bf(f[2], f[3]);
    w[2] = pack2bf(f[4], f[5]);
    w[3] = pack2bf(f[6], f[7]);
    *dst = w;
    if (lane == 0) rn[row] = rsqrtf(ss);
}

// --------------------------------------------------------------------------
// K2: gather anchors, L2-normalize in f32, store bf16; compute Dm.
// One wave per anchor.
// --------------------------------------------------------------------------
__global__ __launch_bounds__(128) void k_anchors(
    const float* __restrict__ out1, const float* __restrict__ out2,
    const int* __restrict__ links,
    const float* __restrict__ rn1, const float* __restrict__ rn2,
    unsigned short* __restrict__ e1n, unsigned short* __restrict__ e2n,
    float* __restrict__ dm, int A, int apad)
{
    int i    = blockIdx.x * 4 + (threadIdx.x >> 5);
    int lane = threadIdx.x & 31;
    if (i >= apad) return;
    u32x4* d1 = (u32x4*)(e1n + (size_t)i * DIMS + lane * 8);
    u32x4* d2 = (u32x4*)(e2n + (size_t)i * DIMS + lane * 8);
    if (i >= A) {
        u32x4 z = {0u, 0u, 0u, 0u};
        *d1 = z; *d2 = z;
        if (lane == 0) dm[i] = 0.0f;
        return;
    }
    int a1 = links[2 * i + 0];
    int a2 = links[2 * i + 1];
    const float* p1 = out1 + (size_t)a1 * DIMS + lane * 8;
    const float* p2 = out2 + (size_t)a2 * DIMS + lane * 8;
    float f1[8], f2[8];
    #pragma unroll
    for (int j = 0; j < 8; ++j) { f1[j] = p1[j]; f2[j] = p2[j]; }
    float dot = 0.0f;
    #pragma unroll
    for (int j = 0; j < 8; ++j) dot += f1[j] * f2[j];
    #pragma unroll
    for (int m = 16; m >= 1; m >>= 1) dot += __shfl_xor(dot, m, 32);
    float r1 = rn1[a1], r2 = rn2[a2];
    if (lane == 0) dm[i] = 1.0f + MARGIN_C - dot * r1 * r2;   // cos_dist + margin
    u32x4 w1, w2;
    #pragma unroll
    for (int j = 0; j < 4; ++j) {
        w1[j] = pack2bf(f1[2 * j] * r1, f1[2 * j + 1] * r1);
        w2[j] = pack2bf(f2[2 * j] * r2, f2[2 * j + 1] * r2);
    }
    *d1 = w1; *d2 = w2;
}

// --------------------------------------------------------------------------
// K3: bf16 WMMA GEMM (16-anchor tile x 16-row N-tiles, K=256) with streaming
// per-anchor top-50.  grid = (NPART/4, apad/16, 2 sides), 4 waves/block,
// wave p of 32 owns N-tiles t = p, p+32, ...
// A sched_barrier between the B-load phase and the WMMA chain forces the
// backend to issue all 16 b128 loads as one clause (distinct registers),
// then wait once and run the 8 WMMAs back-to-back.
// --------------------------------------------------------------------------
__global__ __launch_bounds__(128) void k_gemm_topk(
    const unsigned short* __restrict__ e1n, const unsigned short* __restrict__ e2n,
    const unsigned short* __restrict__ o1b, const unsigned short* __restrict__ o2b,
    const float* __restrict__ rn1, const float* __restrict__ rn2,
    float* __restrict__ cand, int NT, int apad)
{
    const int side = blockIdx.z;
    const unsigned short* E  = side ? e2n : e1n;   // normalized anchor rows (bf16)
    const unsigned short* O  = side ? o1b : o2b;   // un-normalized matrix (bf16)
    const float*          RN = side ? rn1 : rn2;   // f32 inverse norms of O rows

    const int waveId = threadIdx.x >> 5;
    const int lane   = threadIdx.x & 31;
    const int p      = blockIdx.x * 4 + waveId;    // N-partition 0..31
    const int h      = lane >> 4;
    const int l15    = lane & 15;
    const int atile  = blockIdx.y;

    __shared__ float buf[4][16][TK];     // per-wave, per-anchor top-50
    __shared__ float slds[4][16][16];    // per-wave s-tile transpose staging

    if (lane < 16) {
        #pragma unroll
        for (int k = 0; k < TK; ++k) buf[waveId][lane][k] = -3.0e38f;
    }
    float curmin = -3.0e38f;

    // Preload A fragments: 16x256 anchor tile, 8 k-steps of 32.
    // ISA 16-bit A layout: lane&15 = M row; elems 0..7 -> K = k0+8h+e,
    // elems 8..15 -> K = k0+16+8h+(e-8).  Both runs are contiguous 16B.
    v16bf afrag[8];
    {
        const unsigned short* erow = E + ((size_t)atile * 16 + l15) * DIMS;
        #pragma unroll
        for (int ks = 0; ks < 8; ++ks) {
            union { u32x4 q[2]; v16bf v; } ua;
            ua.q[0] = *(const u32x4*)(erow + ks * 32 + 8 * h);
            ua.q[1] = *(const u32x4*)(erow + ks * 32 + 16 + 8 * h);
            afrag[ks] = ua.v;
        }
    }

    const int ntt = (NT + NPART - 1) / NPART;   // uniform trip count for syncthreads
    for (int it = 0; it < ntt; ++it) {
        const int t = p + it * NPART;
        const bool act = (t < NT);              // uniform per wave -> EXEC all-1 at WMMA
        if (act) {
            // ISA 16-bit B layout: lane&15 = N col; elems 0..7 -> K = k0+8h+e, elems 8..15 -> K = k0+16+8h+(e-8).
            const unsigned short* orow = O + ((size_t)t * 16 + l15) * DIMS + 8 * h;
            const float rnv = RN[t * 16 + l15];          // same N col for all 8 acc rows

            // Stage all B fragments: 16 x b128 loads, one clause.
            union { u32x4 q[2]; v16bf v; } ub[8];
            #pragma unroll
            for (int ks = 0; ks < 8; ++ks) {
                const u32x4* pb = (const u32x4*)(orow + ks * 32);
                ub[ks].q[0] = pb[0];
                ub[ks].q[1] = pb[2];
            }

            // Forbid the scheduler from sinking the loads into the WMMA chain.
            __builtin_amdgcn_sched_barrier(0);

            // 8 chained WMMAs, single preceding memory wait.
            v8f c = {0.f, 0.f, 0.f, 0.f, 0.f, 0.f, 0.f, 0.f};
            #pragma unroll
            for (int ks = 0; ks < 8; ++ks) {
                c = __builtin_amdgcn_wmma_f32_16x16x32_bf16(
                        false, afrag[ks], false, ub[ks].v, (short)0, c, false, false);
            }

            #pragma unroll
            for (int r = 0; r < 8; ++r)                  // D: elem r -> M = r + 8h, N = l15
                slds[waveId][r + 8 * h][l15] = c[r] * rnv;
        }
        __syncthreads();
        if (act && lane < 16) {                          // lane l owns anchor row l
            const float* sl = slds[waveId][lane];
            float* bb = buf[waveId][lane];
            #pragma unroll 1
            for (int j = 0; j < 16; ++j) {
                float v = sl[j];
                if (v > curmin) {                        // rare after warm-up
                    int mi = 0; float mv = bb[0];
                    #pragma unroll 1
                    for (int k = 1; k < TK; ++k) { float x = bb[k]; if (x < mv) { mv = x; mi = k; } }
                    bb[mi] = v;
                    mv = bb[0];
                    #pragma unroll 1
                    for (int k = 1; k < TK; ++k) mv = fminf(mv, bb[k]);
                    curmin = mv;
                }
            }
        }
        __syncthreads();
    }

    if (lane < 16) {
        float* dst = cand + (((size_t)side * apad + atile * 16 + lane) * NPART + p) * TK;
        #pragma unroll 1
        for (int k = 0; k < TK; ++k) dst[k] = buf[waveId][lane][k];
    }
}

// --------------------------------------------------------------------------
// K4: per anchor, merge 32 partitions x 50 candidates -> global top-50,
// accumulate relu(Dm - 1 + s).  One block per (anchor, side).
// --------------------------------------------------------------------------
__global__ __launch_bounds__(256) void k_loss(
    const float* __restrict__ cand, const float* __restrict__ dm,
    float* __restrict__ out, int apad, float invAK)
{
    const int side = blockIdx.y;
    const int i    = blockIdx.x;
    const int tid  = threadIdx.x;

    __shared__ float vals[NPART * TK];
    __shared__ float rv[256];
    __shared__ int   ri[256];

    const float* src = cand + ((size_t)side * apad + i) * (NPART * TK);
    for (int j = tid; j < NPART * TK; j += 256) vals[j] = src[j];
    __syncthreads();

    const float dmi = dm[i];
    float total = 0.0f;
    for (int it = 0; it < TK; ++it) {
        float bm = -3.4e38f; int bi = 0;
        for (int j = tid; j < NPART * TK; j += 256) {
            float v = vals[j];
            if (v > bm) { bm = v; bi = j; }
        }
        rv[tid] = bm; ri[tid] = bi;
        __syncthreads();
        for (int s2 = 128; s2 > 0; s2 >>= 1) {
            if (tid < s2 && rv[tid + s2] > rv[tid]) { rv[tid] = rv[tid + s2]; ri[tid] = ri[tid + s2]; }
            __syncthreads();
        }
        if (tid == 0) {
            total += fmaxf(dmi - 1.0f + rv[0], 0.0f);
            vals[ri[0]] = -3.4e38f;
        }
        __syncthreads();
    }
    if (tid == 0) atomicAdd(out, total * invAK);
}

__global__ void k_zero(float* out) { if (threadIdx.x == 0) out[0] = 0.0f; }

// --------------------------------------------------------------------------
extern "C" void kernel_launch(void* const* d_in, const int* in_sizes, int n_in,
                              void* d_out, int out_size, void* d_ws, size_t ws_size,
                              hipStream_t stream)
{
    (void)n_in; (void)out_size; (void)ws_size;
    const float* out1  = (const float*)d_in[0];
    const float* out2  = (const float*)d_in[1];
    const int*   links = (const int*)d_in[2];

    const int N    = in_sizes[0] / DIMS;        // 100000
    const int A    = in_sizes[2] / 2;           // 2000
    const int npad = (N + 15) & ~15;
    const int apad = (A + 15) & ~15;

    char* ws = (char*)d_ws;
    size_t off = 0;
    auto alloc = [&](size_t bytes) -> void* {
        void* ptr = ws + off;
        off = (off + bytes + 255) & ~(size_t)255;
        return ptr;
    };
    unsigned short* o1b = (unsigned short*)alloc((size_t)npad * DIMS * 2);
    unsigned short* o2b = (unsigned short*)alloc((size_t)npad * DIMS * 2);
    float* rn1 = (float*)alloc((size_t)npad * 4);
    float* rn2 = (float*)alloc((size_t)npad * 4);
    unsigned short* e1n = (unsigned short*)alloc((size_t)apad * DIMS * 2);
    unsigned short* e2n = (unsigned short*)alloc((size_t)apad * DIMS * 2);
    float* dm   = (float*)alloc((size_t)apad * 4);
    float* cand = (float*)alloc((size_t)2 * apad * NPART * TK * 4);

    k_norm_convert<<<(npad + 7) / 8, 256, 0, stream>>>(out1, o1b, rn1, N, npad);
    k_norm_convert<<<(npad + 7) / 8, 256, 0, stream>>>(out2, o2b, rn2, N, npad);
    k_anchors<<<(apad + 3) / 4, 128, 0, stream>>>(out1, out2, links, rn1, rn2,
                                                  e1n, e2n, dm, A, apad);
    const int NT = npad / 16;
    dim3 g3(NPART / 4, apad / 16, 2);
    k_gemm_topk<<<g3, 128, 0, stream>>>(e1n, e2n, o1b, o2b, rn1, rn2, cand, NT, apad);

    k_zero<<<1, 32, 0, stream>>>((float*)d_out);
    dim3 g4(A, 2);
    k_loss<<<g4, 256, 0, stream>>>(cand, dm, (float*)d_out, apad,
                                   1.0f / ((float)A * (float)TK));
}
